// SutraV053Baseline_34084860461134
// MI455X (gfx1250) — compile-verified, matrix operand order checked by reading, not run on hardware
//
#include <hip/hip_runtime.h>

#define DIMD   512
#define FFD    2048
#define SST    6
#define BB     4
#define TT     512
#define NTOK   (BB * TT)          // 2048
#define WIN    64
#define KRET   8
#define NSCR   8
#define VOCABN 32000
#define NEGF   (-1000000000.0f)
#define INV_SQRT_D 0.044194173824159216f   // 1/sqrt(512)

typedef __attribute__((ext_vector_type(16))) _Float16 v16h;
typedef __attribute__((ext_vector_type(8)))  _Float16 v8h;
typedef __attribute__((ext_vector_type(8)))  float    v8f;

union Frag { v16h v; v8h h[2]; };

__device__ __forceinline__ float siluf(float x)     { return x / (1.0f + __expf(-x)); }
__device__ __forceinline__ float softplusf(float x) { return x > 20.0f ? x : log1pf(__expf(x)); }

// ---------------------------------------------------------------------------
// f32 -> f16 weight conversion
// ---------------------------------------------------------------------------
__global__ void k_cvt(const float* __restrict__ s, _Float16* __restrict__ d, size_t n) {
    size_t i = (size_t)blockIdx.x * blockDim.x + threadIdx.x;
    if (i < n) d[i] = (_Float16)s[i];
}

// ---------------------------------------------------------------------------
// WMMA GEMM: C[M,N] (f32, + optional f16 mirror) = epi(A[M,K] * B)
// A f16 row-major; B f16 [K,N] (or [N,K] if TRANSB).
// 128x64 tile / block, 8 waves, each wave 32x32 (4 accumulators, 4 WMMA/K-step)
// Double-buffered LDS; A tile (and B tile when TRANSB) staged with
// global_load_async_to_lds_b128 (ASYNCcnt), one barrier per K-step.
// ---------------------------------------------------------------------------
template <bool TRANSB>
__global__ __launch_bounds__(256) void k_gemm(
    const _Float16* __restrict__ A, const _Float16* __restrict__ B,
    float* __restrict__ C, _Float16* __restrict__ C16,
    const float* __restrict__ bias,
    int M, int N, int K, int epi, float alpha, float addc)
{
    __shared__ _Float16 As[2 * 128 * 40];   // [buf][row][k], stride 40 halves
    __shared__ _Float16 Bs[2 * 64 * 40];    // [buf][col][k], stride 40 halves

    const int tid  = threadIdx.x;
    const int m0   = blockIdx.y * 128;
    const int n0   = blockIdx.x * 64;
    const int wave = tid >> 5, lane = tid & 31;
    const int mq   = wave & 3;     // 32-row slice 0..3
    const int np   = wave >> 2;    // 32-col slice 0..1
    const int r    = lane & 15, hsel = lane >> 4;

    v8f acc00 = {}, acc01 = {}, acc10 = {}, acc11 = {};

    const bool fastM = (m0 + 128 <= M);

    auto stage = [&](int k0, int bsel) {
        _Float16* ab = As + bsel * (128 * 40);
        _Float16* bb = Bs + bsel * (64 * 40);
        // ---- A tile: 128 rows x 32 k (8KB), 16B per lane per op ----
        if (fastM) {
            #pragma unroll
            for (int c = tid; c < 512; c += 256) {
                int row = c >> 2, kg = (c & 3) * 8;
                const _Float16* gp = A + (size_t)(m0 + row) * K + k0 + kg;
                unsigned lds = (unsigned)(size_t)(ab + row * 40 + kg);
                asm volatile("global_load_async_to_lds_b128 %0, %1, off"
                             :: "v"(lds), "v"(gp) : "memory");
            }
        } else {
            for (int c = tid; c < 512; c += 256) {
                int row = c >> 2, kg = (c & 3) * 8;
                int gr = m0 + row;
                #pragma unroll
                for (int j = 0; j < 8; ++j)
                    ab[row * 40 + kg + j] =
                        (gr < M) ? A[(size_t)gr * K + k0 + kg + j] : (_Float16)0.f;
            }
        }
        // ---- B tile: 32 k x 64 cols, stored col(n)-major ----
        if (TRANSB) {
            int nn = tid >> 2, kg = (tid & 3) * 8;
            int gn = n0 + nn;
            if (gn < N) {
                const _Float16* gp = B + (size_t)gn * K + k0 + kg;
                unsigned lds = (unsigned)(size_t)(bb + nn * 40 + kg);
                asm volatile("global_load_async_to_lds_b128 %0, %1, off"
                             :: "v"(lds), "v"(gp) : "memory");
            } else {
                #pragma unroll
                for (int j = 0; j < 8; ++j) bb[nn * 40 + kg + j] = (_Float16)0.f;
            }
        } else {
            int kk = tid >> 3, ng = (tid & 7) * 8;
            if (n0 + 64 <= N) {
                v8h bv = *(const v8h*)(B + (size_t)(k0 + kk) * N + n0 + ng);
                #pragma unroll
                for (int j = 0; j < 8; ++j) bb[(ng + j) * 40 + kk] = bv[j];
            } else {
                #pragma unroll
                for (int j = 0; j < 8; ++j) {
                    int gn = n0 + ng + j;
                    bb[(ng + j) * 40 + kk] =
                        (gn < N) ? B[(size_t)(k0 + kk) * N + gn] : (_Float16)0.f;
                }
            }
        }
    };

    stage(0, 0);
    asm volatile("s_wait_asynccnt 0" ::: "memory");
    __syncthreads();

    int buf = 0;
    for (int k0 = 0; k0 < K; k0 += 32) {
        if (k0 + 32 < K) stage(k0 + 32, buf ^ 1);

        const _Float16* ab = As + buf * (128 * 40);
        const _Float16* bb = Bs + buf * (64 * 40);
        Frag a0, a1, b0, b1;
        const int ar0 = (mq * 32 + r) * 40, ar1 = (mq * 32 + 16 + r) * 40;
        a0.h[0] = *(const v8h*)(ab + ar0 + hsel * 8);
        a0.h[1] = *(const v8h*)(ab + ar0 + 16 + hsel * 8);
        a1.h[0] = *(const v8h*)(ab + ar1 + hsel * 8);
        a1.h[1] = *(const v8h*)(ab + ar1 + 16 + hsel * 8);
        const int bc0 = (np * 32 + r) * 40, bc1 = (np * 32 + 16 + r) * 40;
        b0.h[0] = *(const v8h*)(bb + bc0 + hsel * 8);
        b0.h[1] = *(const v8h*)(bb + bc0 + 16 + hsel * 8);
        b1.h[0] = *(const v8h*)(bb + bc1 + hsel * 8);
        b1.h[1] = *(const v8h*)(bb + bc1 + 16 + hsel * 8);

        acc00 = __builtin_amdgcn_wmma_f32_16x16x32_f16(false, a0.v, false, b0.v,
                                                       (short)0, acc00, false, false);
        acc01 = __builtin_amdgcn_wmma_f32_16x16x32_f16(false, a0.v, false, b1.v,
                                                       (short)0, acc01, false, false);
        acc10 = __builtin_amdgcn_wmma_f32_16x16x32_f16(false, a1.v, false, b0.v,
                                                       (short)0, acc10, false, false);
        acc11 = __builtin_amdgcn_wmma_f32_16x16x32_f16(false, a1.v, false, b1.v,
                                                       (short)0, acc11, false, false);

        asm volatile("s_wait_asynccnt 0" ::: "memory");
        __syncthreads();
        buf ^= 1;
    }

    // ---- epilogue + store (f32 + optional f16 mirror) ----
    #pragma unroll
    for (int ai = 0; ai < 2; ++ai) {
        #pragma unroll
        for (int bi = 0; bi < 2; ++bi) {
            v8f acc = ai == 0 ? (bi == 0 ? acc00 : acc01)
                              : (bi == 0 ? acc10 : acc11);
            int c = n0 + np * 32 + bi * 16 + r;
            if (c >= N) continue;
            float bv = (epi >= 1 && bias) ? bias[c] : 0.f;
            #pragma unroll
            for (int i = 0; i < 8; ++i) {
                int rr = m0 + mq * 32 + ai * 16 + hsel * 8 + i;
                if (rr >= M) continue;
                float v = acc[i] + bv;
                if (epi == 2)      v = siluf(v);
                else if (epi == 3) v = softplusf(v) + addc;
                v *= alpha;
                C[(size_t)rr * N + c] = v;
                if (C16) C16[(size_t)rr * N + c] = (_Float16)v;
            }
        }
    }
}

// ---------------------------------------------------------------------------
// glue kernels
// ---------------------------------------------------------------------------
__global__ void k_embed(const int* __restrict__ x, const float* __restrict__ emb,
                        const float* __restrict__ pos, float* __restrict__ h,
                        _Float16* __restrict__ h16) {
    int tok = blockIdx.x;
    int t   = tok % TT;
    int id  = x[tok];
    for (int d = threadIdx.x; d < DIMD; d += blockDim.x) {
        float v = emb[(size_t)id * DIMD + d] + pos[(size_t)t * DIMD + d];
        h[(size_t)tok * DIMD + d]   = v;
        h16[(size_t)tok * DIMD + d] = (_Float16)v;
    }
}

__global__ void k_pi_init(float* __restrict__ pi) {
    int i = blockIdx.x * blockDim.x + threadIdx.x;
    if (i < NTOK * SST) pi[i] = ((i % SST) == 2) ? 1.0f : 0.0f;
}

__global__ void k_mem_init(const float* __restrict__ mi, float* __restrict__ mem) {
    int i = blockIdx.x * blockDim.x + threadIdx.x;
    if (i < BB * NSCR * DIMD) mem[i] = mi[i % (NSCR * DIMD)];
}

__global__ void k_rowmean(const float* __restrict__ mu, float* __restrict__ mub) {
    int d = blockIdx.x * blockDim.x + threadIdx.x;
    int b = blockIdx.y;
    if (d >= DIMD) return;
    float s = 0.f;
    for (int t = 0; t < TT; ++t) s += mu[(size_t)(b * TT + t) * DIMD + d];
    mub[b * DIMD + d] = s * (1.0f / TT);
}

__global__ __launch_bounds__(64) void k_gate(
    const float* __restrict__ mub, const float* __restrict__ gw1, const float* __restrict__ gb1,
    const float* __restrict__ gw2, const float* __restrict__ gb2,
    const float* __restrict__ modeb, float* __restrict__ mode) {
    __shared__ float hid[32];
    __shared__ float mx[2];
    int b = blockIdx.x, j = threadIdx.x;
    if (j < 32) {
        float s = gb1[j];
        for (int d = 0; d < DIMD; ++d) s += mub[b * DIMD + d] * gw1[d * 32 + j];
        hid[j] = siluf(s);
    }
    __syncthreads();
    if (j == 0) {
        float l0 = gb2[0], l1 = gb2[1];
        for (int i = 0; i < 32; ++i) { l0 += hid[i] * gw2[i * 2 + 0]; l1 += hid[i] * gw2[i * 2 + 1]; }
        float m = fmaxf(l0, l1);
        float e0 = __expf(l0 - m), e1 = __expf(l1 - m);
        mx[0] = e0 / (e0 + e1); mx[1] = e1 / (e0 + e1);
    }
    __syncthreads();
    if (j < SST * SST)
        mode[b * 36 + j] = mx[0] * modeb[j] + mx[1] * modeb[36 + j];
}

__global__ void k_pi_ev(const float* __restrict__ base, const float* __restrict__ mode,
                        const float* __restrict__ pi, float* __restrict__ pe) {
    int tok = blockIdx.x * blockDim.x + threadIdx.x;
    if (tok >= NTOK) return;
    int b = tok / TT;
    float Kr[SST][SST];
    for (int i = 0; i < SST; ++i) {
        float row[SST]; float mxv = -1e38f;
        for (int j = 0; j < SST; ++j) {
            int dif = i - j; if (dif < 0) dif = -dif;
            float v = (dif <= 1) ? (base[(size_t)tok * 36 + i * 6 + j] + mode[b * 36 + i * 6 + j])
                                 : NEGF;
            row[j] = v; mxv = fmaxf(mxv, v);
        }
        float sum = 0.f;
        for (int j = 0; j < SST; ++j) { row[j] = __expf(row[j] - mxv); sum += row[j]; }
        float inv = 1.f / sum;
        for (int j = 0; j < SST; ++j) Kr[i][j] = row[j] * inv;
    }
    for (int k2 = 0; k2 < SST; ++k2) {
        float s = 0.f;
        for (int i = 0; i < SST; ++i) s += pi[tok * SST + i] * Kr[i][k2];
        pe[tok * SST + k2] = s;
    }
}

__global__ void k_so_accum(const float* __restrict__ outs, const float* __restrict__ pi,
                           float* __restrict__ so, int s, int first) {
    size_t i = (size_t)blockIdx.x * blockDim.x + threadIdx.x;
    if (i >= (size_t)NTOK * DIMD) return;
    int tok = (int)(i / DIMD);
    float v = pi[tok * SST + s] * outs[i];
    so[i] = first ? v : so[i] + v;
}

__global__ void k_ev_pi(const float* __restrict__ mu, const float* __restrict__ evw,
                        const float* __restrict__ evb, const float* __restrict__ pe,
                        float* __restrict__ pi) {
    int tok = blockIdx.x * blockDim.x + threadIdx.x;
    if (tok >= NTOK) return;
    const float* mp = mu + (size_t)tok * DIMD;
    float ev[SST];
    for (int s = 0; s < SST; ++s) ev[s] = evb[s];
    for (int d = 0; d < DIMD; ++d) {
        float m = mp[d];
        for (int s = 0; s < SST; ++s) ev[s] += m * evw[d * SST + s];
    }
    float mxv = -1e38f;
    for (int s = 0; s < SST; ++s) { ev[s] *= 2.0f; mxv = fmaxf(mxv, ev[s]); }  // /0.5
    float se = 0.f;
    for (int s = 0; s < SST; ++s) { ev[s] = __expf(ev[s] - mxv); se += ev[s]; }
    float p[SST]; float sum = 0.f;
    for (int s = 0; s < SST; ++s) { p[s] = pe[tok * SST + s] * (ev[s] / se); sum += p[s]; }
    sum = fmaxf(sum, 1e-8f);
    for (int s = 0; s < SST; ++s) p[s] /= sum;
    float m1v = -1e38f, m2v = -1e38f;
    for (int s = 0; s < SST; ++s) {
        if (p[s] > m1v) { m2v = m1v; m1v = p[s]; } else if (p[s] > m2v) m2v = p[s];
    }
    float s2 = 0.f;
    for (int s = 0; s < SST; ++s) { p[s] = (p[s] >= m2v) ? p[s] : 0.f; s2 += p[s]; }
    s2 = fmaxf(s2, 1e-8f);
    for (int s = 0; s < SST; ++s) pi[tok * SST + s] = p[s] / s2;
}

__global__ __launch_bounds__(128) void k_attn(const float* __restrict__ Q,
                                              const float* __restrict__ Kx,
                                              const float* __restrict__ V,
                                              _Float16* __restrict__ out16) {
    const int tok = blockIdx.x;
    const int b = tok / TT, t = tok % TT;
    const int s0 = (t >= WIN - 1) ? (t - (WIN - 1)) : 0;
    const int ns = t - s0 + 1;
    __shared__ float qs[DIMD];
    __shared__ float sc[WIN];
    for (int d = threadIdx.x; d < DIMD; d += blockDim.x) qs[d] = Q[(size_t)tok * DIMD + d];
    __syncthreads();
    for (int j = threadIdx.x; j < ns; j += blockDim.x) {
        const float* kp = Kx + (size_t)(b * TT + s0 + j) * DIMD;
        float a = 0.f;
        for (int d = 0; d < DIMD; ++d) a += qs[d] * kp[d];
        sc[j] = a * INV_SQRT_D;
    }
    __syncthreads();
    if (threadIdx.x == 0) {
        float thr = -1e30f;
        if (ns > KRET) {
            float tmp[WIN];
            for (int j = 0; j < ns; ++j) tmp[j] = sc[j];
            for (int it = 0; it < KRET; ++it) {
                int bi = 0; float bvv = tmp[0];
                for (int j = 1; j < ns; ++j) if (tmp[j] > bvv) { bvv = tmp[j]; bi = j; }
                thr = bvv; tmp[bi] = -1e38f;
            }
        }
        float mxv = -1e38f;
        for (int j = 0; j < ns; ++j) if (sc[j] >= thr) mxv = fmaxf(mxv, sc[j]);
        float sum = 0.f;
        for (int j = 0; j < ns; ++j) {
            float w = (sc[j] >= thr) ? __expf(sc[j] - mxv) : 0.f;
            sc[j] = w; sum += w;
        }
        float inv = 1.f / sum;
        for (int j = 0; j < ns; ++j) sc[j] *= inv;
    }
    __syncthreads();
    for (int d = threadIdx.x; d < DIMD; d += blockDim.x) {
        float a = 0.f;
        for (int j = 0; j < ns; ++j) a += sc[j] * V[(size_t)(b * TT + s0 + j) * DIMD + d];
        out16[(size_t)tok * DIMD + d] = (_Float16)a;
    }
}

__global__ __launch_bounds__(64) void k_memread(const float* __restrict__ Qs,
                                                const float* __restrict__ mem,
                                                _Float16* __restrict__ ctx16) {
    const int tok = blockIdx.x, b = tok / TT;
    __shared__ float a[NSCR];
    if (threadIdx.x < NSCR) {
        const float* mp = mem + (size_t)(b * NSCR + threadIdx.x) * DIMD;
        const float* qp = Qs + (size_t)tok * DIMD;
        float s = 0.f;
        for (int d = 0; d < DIMD; ++d) s += qp[d] * mp[d];
        a[threadIdx.x] = s * INV_SQRT_D;
    }
    __syncthreads();
    if (threadIdx.x == 0) {
        float mxv = a[0];
        for (int n = 1; n < NSCR; ++n) mxv = fmaxf(mxv, a[n]);
        float sum = 0.f;
        for (int n = 0; n < NSCR; ++n) { a[n] = __expf(a[n] - mxv); sum += a[n]; }
        for (int n = 0; n < NSCR; ++n) a[n] /= sum;
    }
    __syncthreads();
    for (int d = threadIdx.x; d < DIMD; d += blockDim.x) {
        float s = 0.f;
        for (int n = 0; n < NSCR; ++n) s += a[n] * mem[(size_t)(b * NSCR + n) * DIMD + d];
        ctx16[(size_t)tok * DIMD + d] = (_Float16)s;
    }
}

__global__ void k_msg_combine(const float* __restrict__ m1, const float* __restrict__ m2,
                              const float* __restrict__ pi, _Float16* __restrict__ msg16) {
    size_t i = (size_t)blockIdx.x * blockDim.x + threadIdx.x;
    if (i >= (size_t)NTOK * DIMD) return;
    int tok = (int)(i / DIMD);
    float g = pi[tok * SST + 3];
    msg16[i] = (_Float16)((m1[i] + 0.1f * m2[i]) * g);
}

__global__ void k_bayes(float* __restrict__ mu, _Float16* __restrict__ mu16,
                        float* __restrict__ lam,
                        const float* __restrict__ omu, const float* __restrict__ olam,
                        const float* __restrict__ so, const float* __restrict__ pi) {
    size_t i = (size_t)blockIdx.x * blockDim.x + threadIdx.x;
    if (i >= (size_t)NTOK * DIMD) return;
    int tok = (int)(i / DIMD);
    float g = pi[tok * SST + 4];
    float ol = olam[i];
    float ln_ = lam[i] + g * ol;
    float m = (lam[i] * mu[i] + g * ol * omu[i]) / ln_;
    lam[i] = ln_;
    m += 0.1f * so[i];
    mu[i] = m;
    mu16[i] = (_Float16)m;
}

__global__ __launch_bounds__(64) void k_aw(const float* __restrict__ mu,
                                           const float* __restrict__ mem,
                                           const float* __restrict__ pi,
                                           float* __restrict__ aw) {
    const int tok = blockIdx.x, b = tok / TT;
    __shared__ float a[NSCR];
    if (threadIdx.x < NSCR) {
        const float* mp = mem + (size_t)(b * NSCR + threadIdx.x) * DIMD;
        const float* qp = mu + (size_t)tok * DIMD;
        float s = 0.f;
        for (int d = 0; d < DIMD; ++d) s += qp[d] * mp[d];
        a[threadIdx.x] = s * INV_SQRT_D;
    }
    __syncthreads();
    if (threadIdx.x == 0) {
        float g = pi[tok * SST + 4];
        float mxv = a[0];
        for (int n = 1; n < NSCR; ++n) mxv = fmaxf(mxv, a[n]);
        float sum = 0.f;
        float e[NSCR];
        for (int n = 0; n < NSCR; ++n) { e[n] = __expf(a[n] - mxv); sum += e[n]; }
        for (int n = 0; n < NSCR; ++n) aw[(size_t)tok * NSCR + n] = e[n] / sum * g;
    }
}

__global__ __launch_bounds__(256) void k_memupd(const float* __restrict__ aw,
                                                const float* __restrict__ spv,
                                                float* __restrict__ mem) {
    const int b = blockIdx.x / NSCR, n = blockIdx.x % NSCR;
    __shared__ float denom;
    if (threadIdx.x == 0) {
        float s = 0.f;
        for (int t = 0; t < TT; ++t) s += aw[(size_t)(b * TT + t) * NSCR + n];
        denom = s + 1e-6f;
    }
    __syncthreads();
    for (int d = threadIdx.x; d < DIMD; d += blockDim.x) {
        float s = 0.f;
        for (int t = 0; t < TT; ++t)
            s += aw[(size_t)(b * TT + t) * NSCR + n] * spv[(size_t)(b * TT + t) * DIMD + d];
        mem[(size_t)(b * NSCR + n) * DIMD + d] += 0.1f * s / denom;
    }
}

__global__ __launch_bounds__(256) void k_ln(const float* __restrict__ mu,
                                            const float* __restrict__ g,
                                            const float* __restrict__ bta,
                                            _Float16* __restrict__ out16) {
    int tok = blockIdx.x;
    __shared__ float red[256];
    const float* xp = mu + (size_t)tok * DIMD;
    float s = 0.f;
    for (int d = threadIdx.x; d < DIMD; d += 256) s += xp[d];
    red[threadIdx.x] = s; __syncthreads();
    for (int o = 128; o > 0; o >>= 1) {
        if (threadIdx.x < o) red[threadIdx.x] += red[threadIdx.x + o];
        __syncthreads();
    }
    float mean = red[0] * (1.0f / DIMD);
    __syncthreads();
    float v = 0.f;
    for (int d = threadIdx.x; d < DIMD; d += 256) { float t = xp[d] - mean; v += t * t; }
    red[threadIdx.x] = v; __syncthreads();
    for (int o = 128; o > 0; o >>= 1) {
        if (threadIdx.x < o) red[threadIdx.x] += red[threadIdx.x + o];
        __syncthreads();
    }
    float rstd = rsqrtf(red[0] * (1.0f / DIMD) + 1e-5f);
    for (int d = threadIdx.x; d < DIMD; d += 256)
        out16[(size_t)tok * DIMD + d] = (_Float16)((xp[d] - mean) * rstd * g[d] + bta[d]);
}

// ---------------------------------------------------------------------------
// host side
// ---------------------------------------------------------------------------
static void gemm(hipStream_t st, const _Float16* A, const _Float16* B, float* C,
                 _Float16* C16, const float* bias, int M, int N, int K, int epi,
                 float alpha, float addc, bool transb) {
    dim3 grid((N + 63) / 64, (M + 127) / 128), blk(256);
    if (transb) k_gemm<true><<<grid, blk, 0, st>>>(A, B, C, C16, bias, M, N, K, epi, alpha, addc);
    else        k_gemm<false><<<grid, blk, 0, st>>>(A, B, C, C16, bias, M, N, K, epi, alpha, addc);
}

static void cvt(hipStream_t st, const float* s, _Float16* d, size_t n) {
    k_cvt<<<(int)((n + 255) / 256), 256, 0, st>>>(s, d, n);
}

extern "C" void kernel_launch(void* const* d_in, const int* in_sizes, int n_in,
                              void* d_out, int out_size, void* d_ws, size_t ws_size,
                              hipStream_t stream) {
    (void)in_sizes; (void)n_in; (void)out_size; (void)ws_size;

    const int*   x          = (const int*)d_in[0];
    const float* emb        = (const float*)d_in[1];
    const float* pos_emb    = (const float*)d_in[2];
    const float* init_mu_w  = (const float*)d_in[3];
    const float* init_mu_b  = (const float*)d_in[4];
    const float* init_lam_w = (const float*)d_in[5];
    const float* init_lam_b = (const float*)d_in[6];
    const float* tr_w1      = (const float*)d_in[7];
    const float* tr_b1      = (const float*)d_in[8];
    const float* tr_w2      = (const float*)d_in[9];
    const float* tr_b2      = (const float*)d_in[10];
    const float* tr_gw1     = (const float*)d_in[11];
    const float* tr_gb1     = (const float*)d_in[12];
    const float* tr_gw2     = (const float*)d_in[13];
    const float* tr_gb2     = (const float*)d_in[14];
    const float* tr_modeb   = (const float*)d_in[15];
    const float* sb_w1      = (const float*)d_in[16];
    const float* sb_b1      = (const float*)d_in[17];
    const float* sb_w2      = (const float*)d_in[18];
    const float* sb_b2      = (const float*)d_in[19];
    const float* ev_w       = (const float*)d_in[20];
    const float* ev_b       = (const float*)d_in[21];
    const float* rq         = (const float*)d_in[22];
    const float* rk         = (const float*)d_in[23];
    const float* rv         = (const float*)d_in[24];
    const float* ro         = (const float*)d_in[25];
    const float* wr_mu_w    = (const float*)d_in[26];
    const float* wr_mu_b    = (const float*)d_in[27];
    const float* wr_lam_w   = (const float*)d_in[28];
    const float* wr_lam_b   = (const float*)d_in[29];
    const float* mem_init   = (const float*)d_in[30];
    const float* sp_q       = (const float*)d_in[31];
    const float* sp_v       = (const float*)d_in[32];
    const float* sp_o       = (const float*)d_in[33];
    const float* ln_g       = (const float*)d_in[34];
    const float* ln_b       = (const float*)d_in[35];
    float* logits = (float*)d_out;

    // ---- workspace carve-out ----
    size_t off = 0;
    auto take = [&](size_t bytes) -> void* {
        void* p = (char*)d_ws + off;
        off += (bytes + 255) & ~(size_t)255;
        return p;
    };
    const size_t DD = (size_t)DIMD * DIMD;
    _Float16* w_imu  = (_Float16*)take(DD * 2);
    _Float16* w_ilam = (_Float16*)take(DD * 2);
    _Float16* w_tr1  = (_Float16*)take((size_t)DIMD * 2 * DIMD * 2);
    _Float16* w_tr2  = (_Float16*)take((size_t)2 * DIMD * 36 * 2);
    _Float16* w_sb1  = (_Float16*)take((size_t)SST * DIMD * FFD * 2);
    _Float16* w_sb2  = (_Float16*)take((size_t)SST * FFD * DIMD * 2);
    _Float16* w_rq   = (_Float16*)take(DD * 2);
    _Float16* w_rk   = (_Float16*)take(DD * 2);
    _Float16* w_rv   = (_Float16*)take(DD * 2);
    _Float16* w_ro   = (_Float16*)take(DD * 2);
    _Float16* w_wmu  = (_Float16*)take(DD * 2);
    _Float16* w_wlam = (_Float16*)take(DD * 2);
    _Float16* w_spq  = (_Float16*)take(DD * 2);
    _Float16* w_spv  = (_Float16*)take(DD * 2);
    _Float16* w_spo  = (_Float16*)take(DD * 2);
    _Float16* w_emb  = (_Float16*)take((size_t)VOCABN * DIMD * 2);

    const size_t TD = (size_t)NTOK * DIMD;
    float* mu      = (float*)take(TD * 4);
    float* lam     = (float*)take(TD * 4);
    float* hbuf    = (float*)take(TD * 4);                 // embedding h (f32)
    float* hid     = (float*)take((size_t)NTOK * FFD * 4); // f32 hidden / m1,m2
    float* outs    = (float*)take(TD * 4);
    float* so      = (float*)take(TD * 4);
    float* qbuf    = (float*)take(TD * 4);
    float* kbuf    = (float*)take(TD * 4);                 // k -> obs_mu
    float* vbuf    = (float*)take(TD * 4);                 // v -> obs_lam
    float* spbuf   = (float*)take(TD * 4);                 // sp_q out -> sp_v out
    float* base    = (float*)take((size_t)NTOK * 36 * 4);
    float* pi      = (float*)take((size_t)NTOK * SST * 4);
    float* pi_ev   = (float*)take((size_t)NTOK * SST * 4);
    float* mode    = (float*)take((size_t)BB * 36 * 4);
    float* mub     = (float*)take((size_t)BB * DIMD * 4);
    float* mem     = (float*)take((size_t)BB * NSCR * DIMD * 4);
    float* aw      = (float*)take((size_t)NTOK * NSCR * 4);
    // f16 activation mirrors (A operands of WMMA GEMMs)
    _Float16* mu16  = (_Float16*)take(TD * 2);
    _Float16* h16   = (_Float16*)take(TD * 2);             // h, then attnv
    _Float16* hid16 = (_Float16*)take((size_t)NTOK * FFD * 2);
    _Float16* ctx16 = (_Float16*)take(TD * 2);
    _Float16* msg16 = (_Float16*)take(TD * 2);
    _Float16* ln16  = (_Float16*)take(TD * 2);
    float* m1 = hid;
    float* m2 = hid + TD;

    // ---- weight conversion (weights fit in the 192MB L2; converted once/launch) ----
    cvt(stream, init_mu_w,  w_imu,  DD);
    cvt(stream, init_lam_w, w_ilam, DD);
    cvt(stream, tr_w1,      w_tr1,  (size_t)DIMD * 2 * DIMD);
    cvt(stream, tr_w2,      w_tr2,  (size_t)2 * DIMD * 36);
    cvt(stream, sb_w1,      w_sb1,  (size_t)SST * DIMD * FFD);
    cvt(stream, sb_w2,      w_sb2,  (size_t)SST * FFD * DIMD);
    cvt(stream, rq,         w_rq,   DD);
    cvt(stream, rk,         w_rk,   DD);
    cvt(stream, rv,         w_rv,   DD);
    cvt(stream, ro,         w_ro,   DD);
    cvt(stream, wr_mu_w,    w_wmu,  DD);
    cvt(stream, wr_lam_w,   w_wlam, DD);
    cvt(stream, sp_q,       w_spq,  DD);
    cvt(stream, sp_v,       w_spv,  DD);
    cvt(stream, sp_o,       w_spo,  DD);
    cvt(stream, emb,        w_emb,  (size_t)VOCABN * DIMD);

    // ---- init ----
    k_embed<<<NTOK, 256, 0, stream>>>(x, emb, pos_emb, hbuf, h16);
    gemm(stream, h16, w_imu,  mu,  mu16,    init_mu_b,  NTOK, DIMD, DIMD, 1, 1.f, 0.f,  false);
    gemm(stream, h16, w_ilam, lam, nullptr, init_lam_b, NTOK, DIMD, DIMD, 3, 1.f, 0.1f, false);
    k_pi_init<<<(NTOK * SST + 255) / 256, 256, 0, stream>>>(pi);
    k_mem_init<<<(BB * NSCR * DIMD + 255) / 256, 256, 0, stream>>>(mem_init, mem);

    const int ETOK = (int)((TD + 255) / 256);

    for (int step = 0; step < 6; ++step) {
        // transition matrix
        gemm(stream, mu16,  w_tr1, hid,  hid16,   tr_b1, NTOK, 2 * DIMD, DIMD,     2, 1.f, 0.f, false);
        gemm(stream, hid16, w_tr2, base, nullptr, tr_b2, NTOK, 36,       2 * DIMD, 1, 1.f, 0.f, false);
        k_rowmean<<<dim3((DIMD + 255) / 256, BB), 256, 0, stream>>>(mu, mub);
        k_gate<<<BB, 64, 0, stream>>>(mub, tr_gw1, tr_gb1, tr_gw2, tr_gb2, tr_modeb, mode);
        k_pi_ev<<<(NTOK + 255) / 256, 256, 0, stream>>>(base, mode, pi, pi_ev);

        // stage blocks (with OLD pi)
        for (int s = 0; s < SST; ++s) {
            gemm(stream, mu16, w_sb1 + (size_t)s * DIMD * FFD, hid, hid16, sb_b1 + s * FFD,
                 NTOK, FFD, DIMD, 2, 1.f, 0.f, false);
            gemm(stream, hid16, w_sb2 + (size_t)s * FFD * DIMD, outs, nullptr, sb_b2 + s * DIMD,
                 NTOK, DIMD, FFD, 1, 1.f, 0.f, false);
            k_so_accum<<<ETOK, 256, 0, stream>>>(outs, pi, so, s, s == 0 ? 1 : 0);
        }

        // pi update (evidence + top2)
        k_ev_pi<<<(NTOK + 255) / 256, 256, 0, stream>>>(mu, ev_w, ev_b, pi_ev, pi);

        // windowed top-k attention
        gemm(stream, mu16, w_rq, qbuf, nullptr, nullptr, NTOK, DIMD, DIMD, 0, 1.f, 0.f, false);
        gemm(stream, mu16, w_rk, kbuf, nullptr, nullptr, NTOK, DIMD, DIMD, 0, 1.f, 0.f, false);
        gemm(stream, mu16, w_rv, vbuf, nullptr, nullptr, NTOK, DIMD, DIMD, 0, 1.f, 0.f, false);
        k_attn<<<NTOK, 128, 0, stream>>>(qbuf, kbuf, vbuf, h16);    // h16 = attn@v (f16)

        // memory read
        gemm(stream, mu16, w_spq, spbuf, nullptr, nullptr, NTOK, DIMD, DIMD, 0, 1.f, 0.f, false);
        k_memread<<<NTOK, 64, 0, stream>>>(spbuf, mem, ctx16);

        // messages = (attnv@ro + 0.1*ctxraw@sp_o) * pi[:,3]
        gemm(stream, h16,   w_ro,  m1, nullptr, nullptr, NTOK, DIMD, DIMD, 0, 1.f, 0.f, false);
        gemm(stream, ctx16, w_spo, m2, nullptr, nullptr, NTOK, DIMD, DIMD, 0, 1.f, 0.f, false);
        k_msg_combine<<<ETOK, 256, 0, stream>>>(m1, m2, pi, msg16);

        // Bayesian update
        gemm(stream, msg16, w_wmu,  kbuf, nullptr, wr_mu_b,  NTOK, DIMD, DIMD, 1, 1.f, 0.f, false);
        gemm(stream, msg16, w_wlam, vbuf, nullptr, wr_lam_b, NTOK, DIMD, DIMD, 3, 1.f, 0.f, false);
        k_bayes<<<ETOK, 256, 0, stream>>>(mu, mu16, lam, kbuf, vbuf, so, pi);

        // memory write (uses updated mu)
        gemm(stream, mu16, w_spv, spbuf, nullptr, nullptr, NTOK, DIMD, DIMD, 0, 1.f, 0.f, false);
        k_aw<<<NTOK, 64, 0, stream>>>(mu, mem, pi, aw);
        k_memupd<<<BB * NSCR, 256, 0, stream>>>(aw, spbuf, mem);
    }

    // ---- final logits ----
    k_ln<<<NTOK, 256, 0, stream>>>(mu, ln_g, ln_b, ln16);
    gemm(stream, ln16, w_emb, logits, nullptr, nullptr, NTOK, VOCABN, DIMD, 0, INV_SQRT_D, 0.f, true);
}
